// ISAB_31885837205792
// MI455X (gfx1250) — matrix operand order repaired and checked by reading, not compiled, not run on hardware
//
#include <hip/hip_runtime.h>

// ---------------------------------------------------------------------------
// ISAB (Set Transformer induced-set attention) for gfx1250 / MI455X.
// All 64x64xK matmuls on V_WMMA_F32_16X16X32_F16 (f16 in, f32 acc).
//
// Pipeline:
//   prep        : seg-offset prefix sum, f32->f16 weights, Q0=(I@WQ0^T)/Z
//   ab0_partial : flash-attention partials over key chunks, J blocks/segment,
//                 double-buffered LDS staging of X (loads overlap WMMA)
//   ab0_reduce  : merge partials, LN+FC(WMMA)+LN -> i_seg(f16);
//                 also precompute K1 = iseg@WK1^T and V1^T once per segment
//   ab1         : per 64-token tile: Q(WMMA), 64-key softmax, O(WMMA),
//                 LN+FC(WMMA)+LN, scatter to ragged output
// ---------------------------------------------------------------------------

typedef __attribute__((ext_vector_type(16))) _Float16 v16h;
typedef __attribute__((ext_vector_type(8)))  _Float16 v8h;
typedef __attribute__((ext_vector_type(8)))  float    v8f;

#define DDIM 64
#define INVZ 0.125f   // 1/sqrt(64)
#define LNEPS 1e-5f

// ---- WMMA wrapper ----------------------------------------------------------
__device__ __forceinline__ v8f wmma_f16(v16h a, v16h b, v8f c) {
  return __builtin_amdgcn_wmma_f32_16x16x32_f16(
      false, a, false, b, (short)0, c, false, false);
}

// ---- Fragment helpers (wave32; A/B use contiguous k, C/D per ISA 7.12.2) -------------------------
// A 16x32 f16: lane holds row (lane&15); halves 0..15 at K=kc+(hi?16:0)+i
// (contiguous, matching ldB).  mat row-major [rows][ld].
__device__ __forceinline__ v16h ldA(const _Float16* mat, int row0, int ld, int kc) {
  int lane = threadIdx.x & 31;
  int r  = row0 + (lane & 15);
  int k0 = kc + ((lane & 16) ? 16 : 0);
  int k1 = k0 + 8;
  v8h a0 = *(const v8h*)(mat + r * ld + k0);
  v8h a1 = *(const v8h*)(mat + r * ld + k1);
  v16h a;
#pragma unroll
  for (int i = 0; i < 8; ++i) { a[i] = a0[i]; a[8 + i] = a1[i]; }
  return a;
}

// B 32x16 f16 where B[k][n] = mat[n][k] (mat row-major [N][K]):
// lane = column n0+(lane&15), 16 consecutive k -> one 32B read.
__device__ __forceinline__ v16h ldB(const _Float16* mat, int n0, int ld, int kc) {
  int lane = threadIdx.x & 31;
  int n = n0 + (lane & 15);
  int k = kc + ((lane & 16) ? 16 : 0);
  return *(const v16h*)(mat + n * ld + k);
}

// C/D f32 frag: vgpr r, lane l -> (row0 + r + (l>=16?8:0), col0 + (l&15))
__device__ __forceinline__ void stD_f16(_Float16* mat, int row0, int col0, int ld, v8f d) {
  int lane  = threadIdx.x & 31;
  int rbase = row0 + ((lane & 16) ? 8 : 0);
  int c     = col0 + (lane & 15);
#pragma unroll
  for (int r = 0; r < 8; ++r) mat[(rbase + r) * ld + c] = (_Float16)d[r];
}
__device__ __forceinline__ void stD_f16T(_Float16* mat, int row0, int col0, int ld, v8f d) {
  int lane  = threadIdx.x & 31;
  int rbase = row0 + ((lane & 16) ? 8 : 0);
  int c     = col0 + (lane & 15);
#pragma unroll
  for (int r = 0; r < 8; ++r) mat[c * ld + rbase + r] = (_Float16)d[r];
}

// ---- row reductions: a C/D row lives in one 16-lane half -------------------
__device__ __forceinline__ float rowmax16(float v) {
#pragma unroll
  for (int m = 1; m < 16; m <<= 1) v = fmaxf(v, __shfl_xor(v, m, 32));
  return v;
}
__device__ __forceinline__ float rowsum16(float v) {
#pragma unroll
  for (int m = 1; m < 16; m <<= 1) v += __shfl_xor(v, m, 32);
  return v;
}

// ---- LayerNorm of a 64-wide row (one thread per row) -----------------------
__device__ __forceinline__ void ln_row(const float* in, float* outf, _Float16* outh,
                                       const float* g, const float* b) {
  float mu = 0.f;
  for (int d = 0; d < DDIM; ++d) mu += in[d];
  mu *= (1.0f / DDIM);
  float var = 0.f;
  for (int d = 0; d < DDIM; ++d) { float t = in[d] - mu; var += t * t; }
  var *= (1.0f / DDIM);
  float rs = rsqrtf(var + LNEPS);
  for (int d = 0; d < DDIM; ++d) {
    float o = (in[d] - mu) * rs * g[d] + b[d];
    if (outf) outf[d] = o;
    if (outh) outh[d] = (_Float16)o;
  }
}

// ===========================================================================
// Kernel 0: prep
// ===========================================================================
__global__ void isab_prep(const int* __restrict__ xlens,
                          const float* __restrict__ I,
                          const float* __restrict__ WQ0, const float* __restrict__ WK0,
                          const float* __restrict__ WV0, const float* __restrict__ FCw0,
                          const float* __restrict__ WQ1, const float* __restrict__ WK1,
                          const float* __restrict__ WV1, const float* __restrict__ FCw1,
                          int* __restrict__ segoffs,
                          _Float16* __restrict__ WK0h, _Float16* __restrict__ WV0h,
                          _Float16* __restrict__ FCw0h,
                          _Float16* __restrict__ WQ1h, _Float16* __restrict__ WK1h,
                          _Float16* __restrict__ WV1h, _Float16* __restrict__ FCw1h,
                          _Float16* __restrict__ Q0h) {
  if (threadIdx.x == 0) {
    int s = 0;
    for (int b = 0; b < 64; ++b) { segoffs[b] = s; s += xlens[b]; }
    segoffs[64] = s;
  }
  for (int i = threadIdx.x; i < 64 * 64; i += blockDim.x) {
    WK0h[i]  = (_Float16)WK0[i];
    WV0h[i]  = (_Float16)WV0[i];
    FCw0h[i] = (_Float16)FCw0[i];
    WQ1h[i]  = (_Float16)WQ1[i];
    WK1h[i]  = (_Float16)WK1[i];
    WV1h[i]  = (_Float16)WV1[i];
    FCw1h[i] = (_Float16)FCw1[i];
  }
  for (int i = threadIdx.x; i < 64 * 64; i += blockDim.x) {
    int m = i >> 6, dq = i & 63;
    float s = 0.f;
    for (int d = 0; d < DDIM; ++d) s += I[m * DDIM + d] * WQ0[dq * DDIM + d];
    Q0h[i] = (_Float16)(s * INVZ);
  }
}

// ===========================================================================
// Kernel 1: AB0 partials.  blockIdx.x = b*J + j; block j handles key chunks
// j, j+J, j+2J, ... with online softmax; writes unnormalized O, m, l.
// X staging is double-buffered: next chunk's global loads are issued before
// the current chunk's WMMAs so VMEM latency overlaps matrix work.
// ===========================================================================
__global__ __launch_bounds__(128) void isab_ab0_partial(
    const float* __restrict__ x, const int* __restrict__ segoffs,
    const int* __restrict__ xlens,
    const _Float16* __restrict__ Q0h, const _Float16* __restrict__ WK0h,
    const _Float16* __restrict__ WV0h,
    float* __restrict__ Op, float* __restrict__ mp, float* __restrict__ lp,
    int J) {
  __shared__ alignas(32) _Float16 Xc[2][64 * 64];
  __shared__ alignas(32) _Float16 Kc[64 * 64];     // K[key][dq]
  __shared__ alignas(32) _Float16 Vt[64 * 64];     // V^T[d][key]
  __shared__ alignas(32) _Float16 Pb[4 * 16 * 64]; // per-wave probs
  __shared__ alignas(32) _Float16 WKs[64 * 64];
  __shared__ alignas(32) _Float16 WVs[64 * 64];

  const int b = blockIdx.x / J;
  const int j = blockIdx.x - b * J;
  const int L   = xlens[b];
  const int off = segoffs[b];
  const int w    = threadIdx.x >> 5;
  const int lane = threadIdx.x & 31;
  const int trow = threadIdx.x >> 6;   // 0/1
  const int tcol = threadIdx.x & 63;

  for (int i = threadIdx.x; i < 64 * 64; i += 128) { WKs[i] = WK0h[i]; WVs[i] = WV0h[i]; }

  v16h qf0 = ldA(Q0h, 16 * w, 64, 0);
  v16h qf1 = ldA(Q0h, 16 * w, 64, 32);

  v8f O[4];
#pragma unroll
  for (int n = 0; n < 4; ++n) O[n] = {};
  float mrun[8], lrun[8];
#pragma unroll
  for (int r = 0; r < 8; ++r) { mrun[r] = -1e30f; lrun[r] = 0.f; }

  const int nch = (L + 63) >> 6;                 // >= 8 >= J
  const int n   = (nch - j + J - 1) / J;         // chunks this block owns

  float regs[32];
  auto loadch = [&](int ch) {
    int kv = min(64, L - ch * 64);
#pragma unroll
    for (int k2 = 0; k2 < 32; ++k2) {
      int row = 2 * k2 + trow;
      regs[k2] = (row < kv) ? x[(size_t)(off + ch * 64 + row) * 64 + tcol] : 0.f;
    }
  };
  auto storech = [&](int buf) {
#pragma unroll
    for (int k2 = 0; k2 < 32; ++k2)
      Xc[buf][(2 * k2 + trow) * 64 + tcol] = (_Float16)regs[k2];
  };

  loadch(j);
  storech(0);
  __syncthreads();

  for (int i = 0; i < n; ++i) {
    const int cur = i & 1;
    const int ch  = j + i * J;
    const int kv  = min(64, L - ch * 64);
    if (i + 1 < n) loadch(j + (i + 1) * J);      // overlap with WMMAs below

    // K = X@WK^T, V = X@WV^T (V stored transposed)
#pragma unroll
    for (int nt = 0; nt < 4; ++nt) {
      v8f aK = {}, aV = {};
#pragma unroll
      for (int c = 0; c < 2; ++c) {
        v16h a = ldA(Xc[cur], 16 * w, 64, 32 * c);
        aK = wmma_f16(a, ldB(WKs, 16 * nt, 64, 32 * c), aK);
        aV = wmma_f16(a, ldB(WVs, 16 * nt, 64, 32 * c), aV);
      }
      stD_f16(Kc, 16 * w, 16 * nt, 64, aK);
      stD_f16T(Vt, 16 * w, 16 * nt, 64, aV);
    }
    __syncthreads();

    // S = Qs @ K^T (scale folded into Q0h)
    v8f S[4];
#pragma unroll
    for (int nt = 0; nt < 4; ++nt) {
      v8f acc = {};
      acc = wmma_f16(qf0, ldB(Kc, 16 * nt, 64, 0), acc);
      acc = wmma_f16(qf1, ldB(Kc, 16 * nt, 64, 32), acc);
      int col = 16 * nt + (lane & 15);
      if (col >= kv) {
#pragma unroll
        for (int r = 0; r < 8; ++r) acc[r] = -1e30f;
      }
      S[nt] = acc;
    }
    // online softmax update
    _Float16* Pw = Pb + w * 16 * 64;
#pragma unroll
    for (int r = 0; r < 8; ++r) {
      float rm = fmaxf(fmaxf(S[0][r], S[1][r]), fmaxf(S[2][r], S[3][r]));
      rm = rowmax16(rm);
      float mnew  = fmaxf(mrun[r], rm);
      float scale = __expf(mrun[r] - mnew);
      float rs = 0.f;
#pragma unroll
      for (int nt = 0; nt < 4; ++nt) {
        float p = __expf(S[nt][r] - mnew);
        S[nt][r] = p;
        rs += p;
      }
      rs = rowsum16(rs);
      lrun[r] = lrun[r] * scale + rs;
      mrun[r] = mnew;
#pragma unroll
      for (int nt = 0; nt < 4; ++nt) O[nt][r] *= scale;
    }
#pragma unroll
    for (int nt = 0; nt < 4; ++nt) stD_f16(Pw, 0, 16 * nt, 64, S[nt]);
    // O += P @ V
#pragma unroll
    for (int nt = 0; nt < 4; ++nt)
#pragma unroll
      for (int c = 0; c < 2; ++c)
        O[nt] = wmma_f16(ldA(Pw, 0, 64, 32 * c), ldB(Vt, 16 * nt, 64, 32 * c), O[nt]);
    __syncthreads();
    if (i + 1 < n) { storech(cur ^ 1); __syncthreads(); }
  }

  // write partials (unnormalized O at scale mrun, plus m, l)
  const size_t pbase = (size_t)(b * J + j) * 64;
  {
    int rbase = 16 * w + ((lane & 16) ? 8 : 0);
#pragma unroll
    for (int nt = 0; nt < 4; ++nt) {
      int c = 16 * nt + (lane & 15);
#pragma unroll
      for (int r = 0; r < 8; ++r) Op[(pbase + rbase + r) * 64 + c] = O[nt][r];
    }
    if ((lane & 15) == 0) {
      int rb = 16 * w + ((lane & 16) ? 8 : 0);
#pragma unroll
      for (int r = 0; r < 8; ++r) {
        mp[pbase + rb + r] = mrun[r];
        lp[pbase + rb + r] = lrun[r];
      }
    }
  }
}

// ===========================================================================
// Kernel 2: AB0 reduce + epilogue.  One block per segment: merge J partials,
// residual + LN + FC(WMMA) + LN -> iseg (f16); then precompute this
// segment's K1 = iseg@WK1^T and V1^T = (iseg@WV1^T)^T once for AB1.
// ===========================================================================
__global__ __launch_bounds__(128) void isab_ab0_reduce(
    const float* __restrict__ I,
    const float* __restrict__ Op, const float* __restrict__ mp,
    const float* __restrict__ lp, int J,
    const _Float16* __restrict__ FCw0h, const float* __restrict__ FCb0,
    const float* __restrict__ g00, const float* __restrict__ b00,
    const float* __restrict__ g01, const float* __restrict__ b01,
    const _Float16* __restrict__ WK1h, const _Float16* __restrict__ WV1h,
    _Float16* __restrict__ iseg, _Float16* __restrict__ K1s,
    _Float16* __restrict__ Vt1s) {
  __shared__ alignas(32) float    Hf[64 * 64];
  __shared__ alignas(32) float    Hn[64 * 64];
  __shared__ alignas(32) _Float16 Hh[64 * 64];
  __shared__ alignas(32) _Float16 Hh2[64 * 64];

  const int b    = blockIdx.x;
  const int w    = threadIdx.x >> 5;
  const int lane = threadIdx.x & 31;

  // merge flash partials: one thread per query row
  if (threadIdx.x < 64) {
    const int q = threadIdx.x;
    const size_t base = (size_t)b * J * 64;
    float m = -1e30f;
    for (int j = 0; j < J; ++j) m = fmaxf(m, mp[base + (size_t)j * 64 + q]);
    float wj[8];
    float l = 0.f;
    for (int j = 0; j < J; ++j) {
      wj[j] = __expf(mp[base + (size_t)j * 64 + q] - m);
      l += lp[base + (size_t)j * 64 + q] * wj[j];
    }
    for (int d = 0; d < 64; ++d) {
      float s = 0.f;
      for (int j = 0; j < J; ++j)
        s += Op[(base + (size_t)j * 64 + q) * 64 + d] * wj[j];
      Hf[q * 64 + d] = s / l + I[q * 64 + d];
    }
  }
  __syncthreads();
  if (threadIdx.x < 64)
    ln_row(&Hf[threadIdx.x * 64], &Hn[threadIdx.x * 64], &Hh[threadIdx.x * 64], g00, b00);
  __syncthreads();
  // FC + residual
  {
    int rbase = 16 * w + ((lane & 16) ? 8 : 0);
#pragma unroll
    for (int nt = 0; nt < 4; ++nt) {
      v8f acc = {};
#pragma unroll
      for (int c = 0; c < 2; ++c)
        acc = wmma_f16(ldA(Hh, 16 * w, 64, 32 * c), ldB(FCw0h, 16 * nt, 64, 32 * c), acc);
      int cc = 16 * nt + (lane & 15);
#pragma unroll
      for (int r = 0; r < 8; ++r) {
        float f = acc[r] + FCb0[cc];
        f = f > 0.f ? f : 0.f;
        Hf[(rbase + r) * 64 + cc] = Hn[(rbase + r) * 64 + cc] + f;
      }
    }
  }
  __syncthreads();
  if (threadIdx.x < 64) {
    ln_row(&Hf[threadIdx.x * 64], nullptr, &Hh2[threadIdx.x * 64], g01, b01);
    for (int d = 0; d < 64; ++d)
      iseg[((size_t)b * 64 + threadIdx.x) * 64 + d] = Hh2[threadIdx.x * 64 + d];
  }
  __syncthreads();
  // precompute K1 and V1^T for AB1 (once per segment instead of per tile)
#pragma unroll
  for (int nt = 0; nt < 4; ++nt) {
    v8f aK = {}, aV = {};
#pragma unroll
    for (int c = 0; c < 2; ++c) {
      v16h a = ldA(Hh2, 16 * w, 64, 32 * c);
      aK = wmma_f16(a, ldB(WK1h, 16 * nt, 64, 32 * c), aK);
      aV = wmma_f16(a, ldB(WV1h, 16 * nt, 64, 32 * c), aV);
    }
    stD_f16(K1s + (size_t)b * 4096, 16 * w, 16 * nt, 64, aK);
    stD_f16T(Vt1s + (size_t)b * 4096, 16 * w, 16 * nt, 64, aV);
  }
}

// ===========================================================================
// Kernel 3: AB1.  One block per (segment, 64-token tile); K1/V1^T preloaded
// from workspace.  Grid = 64x64 tiles; blocks past L exit early.
// ===========================================================================
__global__ __launch_bounds__(128) void isab_ab1(
    const float* __restrict__ x, const int* __restrict__ segoffs,
    const int* __restrict__ xlens,
    const _Float16* __restrict__ K1s, const _Float16* __restrict__ Vt1s,
    const _Float16* __restrict__ WQ1h, const _Float16* __restrict__ FCw1h,
    const float* __restrict__ FCb1,
    const float* __restrict__ g10, const float* __restrict__ b10,
    const float* __restrict__ g11, const float* __restrict__ b11,
    float* __restrict__ out) {
  __shared__ alignas(32) _Float16 Xh[64 * 64];
  __shared__ alignas(32) float    Xf[64 * 64];
  __shared__ alignas(32) _Float16 K1[64 * 64];   // K[m][dq]
  __shared__ alignas(32) _Float16 Vt[64 * 64];   // V^T[d][m]
  __shared__ alignas(32) _Float16 Q1[64 * 64];
  __shared__ alignas(32) _Float16 Pb[4 * 16 * 64];
  __shared__ alignas(32) float    Hf[64 * 64];
  __shared__ alignas(32) float    Hn[64 * 64];
  __shared__ alignas(32) _Float16 Hh[64 * 64];

  const int b  = blockIdx.x >> 6;
  const int t  = blockIdx.x & 63;
  const int L  = xlens[b];
  const int t0 = t * 64;
  if (t0 >= L) return;
  const int nv   = min(64, L - t0);
  const int off  = segoffs[b];
  const int w    = threadIdx.x >> 5;
  const int lane = threadIdx.x & 31;

  for (int i = threadIdx.x; i < 64 * 64; i += 128) {
    int row = i >> 6, col = i & 63;
    float v = (row < nv) ? x[(size_t)(off + t0 + row) * 64 + col] : 0.f;
    Xf[i] = v;
    Xh[i] = (_Float16)v;
    K1[i] = K1s[(size_t)b * 4096 + i];
    Vt[i] = Vt1s[(size_t)b * 4096 + i];
  }
  __syncthreads();

  // Q = (x@WQ1^T)/Z
#pragma unroll
  for (int nt = 0; nt < 4; ++nt) {
    v8f aQ = {};
#pragma unroll
    for (int c = 0; c < 2; ++c)
      aQ = wmma_f16(ldA(Xh, 16 * w, 64, 32 * c), ldB(WQ1h, 16 * nt, 64, 32 * c), aQ);
#pragma unroll
    for (int r = 0; r < 8; ++r) aQ[r] *= INVZ;
    stD_f16(Q1, 16 * w, 16 * nt, 64, aQ);
  }
  __syncthreads();

  // S = Q @ K^T; one-pass softmax over the 64 induced keys
  v8f S[4];
#pragma unroll
  for (int nt = 0; nt < 4; ++nt) {
    v8f acc = {};
#pragma unroll
    for (int c = 0; c < 2; ++c)
      acc = wmma_f16(ldA(Q1, 16 * w, 64, 32 * c), ldB(K1, 16 * nt, 64, 32 * c), acc);
    S[nt] = acc;
  }
  float lsum[8];
#pragma unroll
  for (int r = 0; r < 8; ++r) {
    float rm = fmaxf(fmaxf(S[0][r], S[1][r]), fmaxf(S[2][r], S[3][r]));
    rm = rowmax16(rm);
    float rs = 0.f;
#pragma unroll
    for (int nt = 0; nt < 4; ++nt) {
      float p = __expf(S[nt][r] - rm);
      S[nt][r] = p;
      rs += p;
    }
    lsum[r] = rowsum16(rs);
  }
  _Float16* Pw = Pb + w * 16 * 64;
#pragma unroll
  for (int nt = 0; nt < 4; ++nt) stD_f16(Pw, 0, 16 * nt, 64, S[nt]);

  const int rbase = 16 * w + ((lane & 16) ? 8 : 0);
#pragma unroll
  for (int nt = 0; nt < 4; ++nt) {
    v8f acc = {};
#pragma unroll
    for (int c = 0; c < 2; ++c)
      acc = wmma_f16(ldA(Pw, 0, 64, 32 * c), ldB(Vt, 16 * nt, 64, 32 * c), acc);
    int cc = 16 * nt + (lane & 15);
#pragma unroll
    for (int r = 0; r < 8; ++r)
      Hf[(rbase + r) * 64 + cc] = Xf[(rbase + r) * 64 + cc] + acc[r] / lsum[r];
  }
  __syncthreads();
  if (threadIdx.x < 64)
    ln_row(&Hf[threadIdx.x * 64], &Hn[threadIdx.x * 64], &Hh[threadIdx.x * 64], g10, b10);
  __syncthreads();
#pragma unroll
  for (int nt = 0; nt < 4; ++nt) {
    v8f acc = {};
#pragma unroll
    for (int c = 0; c < 2; ++c)
      acc = wmma_f16(ldA(Hh, 16 * w, 64, 32 * c), ldB(FCw1h, 16 * nt, 64, 32 * c), acc);
    int cc = 16 * nt + (lane & 15);
#pragma unroll
    for (int r = 0; r < 8; ++r) {
      float f = acc[r] + FCb1[cc];
      f = f > 0.f ? f : 0.f;
      Hf[(rbase + r) * 64 + cc] = Hn[(rbase + r) * 64 + cc] + f;
    }
  }
  __syncthreads();
  if (threadIdx.x < 64 && (int)threadIdx.x < nv) {
    const float* in = &Hf[threadIdx.x * 64];
    float mu = 0.f;
    for (int d = 0; d < DDIM; ++d) mu += in[d];
    mu *= (1.0f / DDIM);
    float var = 0.f;
    for (int d = 0; d < DDIM; ++d) { float q = in[d] - mu; var += q * q; }
    var *= (1.0f / DDIM);
    float rstd = rsqrtf(var + LNEPS);
    float* o = &out[(size_t)(off + t0 + threadIdx.x) * 64];
    for (int d = 0; d < DDIM; ++d) o[d] = (in[d] - mu) * rstd * g11[d] + b11[d];
  }
}

// ===========================================================================
extern "C" void kernel_launch(void* const* d_in, const int* in_sizes, int n_in,
                              void* d_out, int out_size, void* d_ws, size_t ws_size,
                              hipStream_t stream) {
  const float* x     = (const float*)d_in[0];
  const int*   xlens = (const int*)  d_in[1];
  const float* I     = (const float*)d_in[2];
  const float* WQ0   = (const float*)d_in[3];
  const float* WK0   = (const float*)d_in[4];
  const float* WV0   = (const float*)d_in[5];
  const float* FCw0  = (const float*)d_in[6];
  const float* FCb0  = (const float*)d_in[7];
  const float* g00   = (const float*)d_in[8];
  const float* b00   = (const float*)d_in[9];
  const float* g01   = (const float*)d_in[10];
  const float* b01   = (const float*)d_in[11];
  const float* WQ1   = (const float*)d_in[12];
  const float* WK1   = (const float*)d_in[13];
  const float* WV1   = (const float*)d_in[14];
  const float* FCw1  = (const float*)d_in[15];
  const float* FCb1  = (const float*)d_in[16];
  const float* g10   = (const float*)d_in[17];
  const float* b10   = (const float*)d_in[18];
  const float* g11   = (const float*)d_in[19];
  const float* b11   = (const float*)d_in[20];
  float* out = (float*)d_out;
  (void)n_in; (void)in_sizes; (void)out_size;

  // Workspace layout:
  //   [0,512)        segoffs (65 x i32)
  //   [512,66048)    f16 weights: WK0,WV0,FCw0,WQ1,WK1,WV1,FCw1,Q0 (8 x 8KB)
  //   [66048,+512K)  iseg  f16 [64][64][64]
  //   next 512K      K1s   f16 [64][64][64]
  //   next 512K      Vt1s  f16 [64][64][64]
  //   then           Op f32 [64][J][64][64], mp/lp f32 [64][J][64]
  char* ws = (char*)d_ws;
  int*      segoffs = (int*)ws;
  _Float16* WK0h  = (_Float16*)(ws + 512 + 0 * 8192);
  _Float16* WV0h  = (_Float16*)(ws + 512 + 1 * 8192);
  _Float16* FCw0h = (_Float16*)(ws + 512 + 2 * 8192);
  _Float16* WQ1h  = (_Float16*)(ws + 512 + 3 * 8192);
  _Float16* WK1h  = (_Float16*)(ws + 512 + 4 * 8192);
  _Float16* WV1h  = (_Float16*)(ws + 512 + 5 * 8192);
  _Float16* FCw1h = (_Float16*)(ws + 512 + 6 * 8192);
  _Float16* Q0h   = (_Float16*)(ws + 512 + 7 * 8192);
  _Float16* iseg  = (_Float16*)(ws + 66048);
  _Float16* K1s   = (_Float16*)(ws + 66048 + 524288);
  _Float16* Vt1s  = (_Float16*)(ws + 66048 + 2 * 524288);
  const size_t fixed = 66048 + 3 * 524288;           // 1,638,912 B

  // pick key-split factor J from available workspace (deterministic)
  int J = 8;
  while (J > 1 && fixed + (size_t)J * (64 * 4096 * 4 + 2 * 64 * 64 * 4) > ws_size)
    J >>= 1;
  float* Op = (float*)(ws + fixed);
  float* mp = (float*)(ws + fixed + (size_t)J * 64 * 4096 * 4);
  float* lp = (float*)(ws + fixed + (size_t)J * 64 * 4096 * 4 + (size_t)J * 64 * 64 * 4);

  isab_prep<<<1, 256, 0, stream>>>(xlens, I, WQ0, WK0, WV0, FCw0, WQ1, WK1, WV1, FCw1,
                                   segoffs, WK0h, WV0h, FCw0h, WQ1h, WK1h, WV1h, FCw1h,
                                   Q0h);
  isab_ab0_partial<<<64 * J, 128, 0, stream>>>(x, segoffs, xlens, Q0h, WK0h, WV0h,
                                               Op, mp, lp, J);
  isab_ab0_reduce<<<64, 128, 0, stream>>>(I, Op, mp, lp, J, FCw0h, FCb0,
                                          g00, b00, g01, b01, WK1h, WV1h,
                                          iseg, K1s, Vt1s);
  isab_ab1<<<64 * 64, 128, 0, stream>>>(x, segoffs, xlens, K1s, Vt1s, WQ1h, FCw1h,
                                        FCb1, g10, b10, g11, b11, out);
}
